// AttnBlock_19146964206263
// MI455X (gfx1250) — matrix operand order repaired and checked by reading, not compiled, not run on hardware
//
#include <hip/hip_runtime.h>

// ---------------------------------------------------------------------------
// CDNA5 / gfx1250 attention block:
//   LN -> QKV gemm -> S = Q^T K -> softmax -> A = V W^T -> proj (+bias+resid)
// GEMMs: v_wmma_f32_16x16x32_bf16, 256x128 block tile, 64x64 wave tile,
// K-step 64, LDS staging via global_load_async_to_lds_b128 where possible.
// ---------------------------------------------------------------------------

typedef __attribute__((ext_vector_type(16))) __bf16 v16bf;
typedef __attribute__((ext_vector_type(8)))  float  v8f;
typedef int v4i_g __attribute__((vector_size(16)));   // matches builtin's V4i param

#define KSTEP     64
#define LDS_PITCH 72   // 64 bf16 data + 8 pad; 144B rows stay 16B aligned
#define BM        256
#define BN        128

#if defined(__has_builtin)
#if __has_builtin(__builtin_amdgcn_global_load_async_to_lds_b128)
#define HAVE_ASYNC_BUILTIN 1
#endif
#if __has_builtin(__builtin_amdgcn_s_wait_asynccnt)
#define HAVE_WAIT_BUILTIN 1
#endif
#endif

__device__ __forceinline__ void async_b128_to_lds(const unsigned short* src, unsigned short* dst) {
#ifdef HAVE_ASYNC_BUILTIN
    __builtin_amdgcn_global_load_async_to_lds_b128(
        (__attribute__((address_space(1))) v4i_g*)src,
        (__attribute__((address_space(3))) v4i_g*)dst, 0, 0);
#else
    unsigned lds_off = (unsigned)(unsigned long long)(uintptr_t)dst;  // addr[31:0] == LDS offset
    asm volatile("global_load_async_to_lds_b128 %0, %1, off"
                 :: "v"(lds_off), "v"((unsigned long long)(uintptr_t)src)
                 : "memory");
#endif
}

__device__ __forceinline__ void wait_async0() {
#ifdef HAVE_WAIT_BUILTIN
    __builtin_amdgcn_s_wait_asynccnt(0);
#else
    asm volatile("s_wait_asynccnt 0" ::: "memory");
#endif
}

__device__ __forceinline__ unsigned short f2bf(float f) {
    unsigned int u = __float_as_uint(f);
    u += 0x7FFFu + ((u >> 16) & 1u);     // round-to-nearest-even
    return (unsigned short)(u >> 16);
}

union FragU { v16bf v; unsigned short u[16]; };

// A fragment (16x32 bf16): lanes0-15 M=lane hold K 0..15,
// lanes16-31 hold K 16..31 (pairs per VGPR).
__device__ __forceinline__ v16bf frag_a_from_lds(const unsigned short* Ts, int mbase, int kk, int lane) {
    FragU r;
    const unsigned short* p = Ts + (mbase + (lane & 15)) * LDS_PITCH + kk;
    int kh = (lane >> 4) << 4;           // 0 or 16
#pragma unroll
    for (int j = 0; j < 8; ++j) {
        r.u[j]     = p[kh + j];
        r.u[8 + j] = p[8 + kh + j];
    }
    return r.v;
}

// B fragment (32x16 bf16): lane = column n; lanes0-15 hold K=0..15,
// lanes16-31 hold K=16..31.
__device__ __forceinline__ v16bf frag_b_from_lds(const unsigned short* Ts, int nbase, int kk, int lane) {
    FragU r;
    const unsigned short* p = Ts + (nbase + (lane & 15)) * LDS_PITCH + kk + ((lane >> 4) << 4);
#pragma unroll
    for (int j = 0; j < 16; ++j) r.u[j] = p[j];
    return r.v;
}

// Stage a rows x KSTEP bf16 tile into LDS. trans==1: global element (row,k)
// lives at G[k*ld + row] (transpose while staging; VGPR path).
__device__ __forceinline__ void tile_to_lds(unsigned short* Ts, const unsigned short* G,
                                            int ld, int rbase, int kb, int trans, int tid, int rows) {
    if (!trans) {
        int chunks = rows * (KSTEP / 8);             // 16B chunks
        for (int i = tid; i < chunks; i += 256) {
            int row = i >> 3;
            int kc  = (i & 7) << 3;
            async_b128_to_lds(G + (size_t)(rbase + row) * ld + kb + kc,
                              Ts + row * LDS_PITCH + kc);
        }
    } else {
        int sh = (rows == 256) ? 5 : 4;              // chunks per k-row
        int cm = (1 << sh) - 1;
        int chunks = KSTEP << sh;
        for (int i = tid; i < chunks; i += 256) {
            int krow = i >> sh;
            int rc   = (i & cm) << 3;
            union { uint4 v; unsigned short u[8]; } t;
            t.v = *(const uint4*)(G + (size_t)(kb + krow) * ld + rbase + rc);
#pragma unroll
            for (int e = 0; e < 8; ++e) Ts[(rc + e) * LDS_PITCH + krow] = t.u[e];
        }
    }
}

// C[m,n] = sum_k A[m,k] * W[n,k]  (+bias[n]) (+resid[m,n])
// 256x128 tile per workgroup, 8 waves (4x2), 64x64 per wave -> 32 WMMA/stage.
__global__ __launch_bounds__(256) void gemm_bf16_kernel(
    const unsigned short* __restrict__ A, int lda, long long sA, int tA,
    const unsigned short* __restrict__ W, int ldw, long long sW, int tW,
    const float* __restrict__ bias,
    const float* __restrict__ resid, long long sR,
    float* __restrict__ outF, unsigned short* __restrict__ outB,
    int ldc, long long sC, int K)
{
    __shared__ unsigned short As[BM * LDS_PITCH];    // 36 KB
    __shared__ unsigned short Bs[BN * LDS_PITCH];    // 18 KB

    const int tid  = threadIdx.x;
    const int lane = tid & 31;
    const int wave = tid >> 5;
    const int wm   = (wave & 3) * 64;
    const int wn   = (wave >> 2) * 64;

    const int nblk = blockIdx.x * BN;
    const int mblk = blockIdx.y * BM;
    const int z    = blockIdx.z;

    const unsigned short* Ab = A + (size_t)z * sA;
    const unsigned short* Wb = W + (size_t)z * sW;

    const v8f zero8 = {0.f, 0.f, 0.f, 0.f, 0.f, 0.f, 0.f, 0.f};
    v8f acc[4][4];
#pragma unroll
    for (int i = 0; i < 4; ++i)
#pragma unroll
        for (int j = 0; j < 4; ++j) acc[i][j] = zero8;

    for (int kb = 0; kb < K; kb += KSTEP) {
        tile_to_lds(As, Ab, lda, mblk, kb, tA, tid, BM);
        tile_to_lds(Bs, Wb, ldw, nblk, kb, tW, tid, BN);
        if (!tA && kb + KSTEP < K)                   // prefetch next A tile
            __builtin_prefetch(Ab + (size_t)(mblk + (tid >> 1)) * lda + kb + KSTEP, 0, 1);
        wait_async0();
        __syncthreads();

#pragma unroll
        for (int kk = 0; kk < KSTEP; kk += 32) {
            v16bf af[4];
#pragma unroll
            for (int i = 0; i < 4; ++i) af[i] = frag_a_from_lds(As, wm + i * 16, kk, lane);
#pragma unroll
            for (int j = 0; j < 4; ++j) {
                v16bf bf = frag_b_from_lds(Bs, wn + j * 16, kk, lane);
#pragma unroll
                for (int i = 0; i < 4; ++i)
                    acc[i][j] = __builtin_amdgcn_wmma_f32_16x16x32_bf16(
                        false, af[i], false, bf, (short)0, acc[i][j], false, false);
            }
        }
        __syncthreads();
    }

    // C/D layout: VGPR r -> (M=r, N=lane) lanes0-15, (M=r+8, N=lane-16) lanes16-31
    const int mofs = (lane < 16) ? 0 : 8;
    const int nl   = lane & 15;
#pragma unroll
    for (int i = 0; i < 4; ++i)
#pragma unroll
        for (int j = 0; j < 4; ++j)
#pragma unroll
            for (int r = 0; r < 8; ++r) {
                int row = mblk + wm + i * 16 + r + mofs;
                int col = nblk + wn + j * 16 + nl;
                float v = acc[i][j][r];
                if (bias)  v += bias[col];
                if (resid) v += resid[(size_t)z * sR + (size_t)row * ldc + col];
                size_t idx = (size_t)z * sC + (size_t)row * ldc + col;
                if (outF) outF[idx] = v;
                if (outB) outB[idx] = f2bf(v);
            }
}

__global__ __launch_bounds__(256) void ln_kernel(
    const float* __restrict__ x, const float* __restrict__ g,
    const float* __restrict__ b, unsigned short* __restrict__ hb, int D)
{
    __shared__ float red[256];
    const float* xr = x + (size_t)blockIdx.x * D;
    float s = 0.f;
    for (int c = threadIdx.x; c < D; c += 256) s += xr[c];
    red[threadIdx.x] = s; __syncthreads();
    for (int o = 128; o > 0; o >>= 1) {
        if (threadIdx.x < o) red[threadIdx.x] += red[threadIdx.x + o];
        __syncthreads();
    }
    float mu = red[0] / D; __syncthreads();
    float s2 = 0.f;
    for (int c = threadIdx.x; c < D; c += 256) { float d = xr[c] - mu; s2 += d * d; }
    red[threadIdx.x] = s2; __syncthreads();
    for (int o = 128; o > 0; o >>= 1) {
        if (threadIdx.x < o) red[threadIdx.x] += red[threadIdx.x + o];
        __syncthreads();
    }
    float rstd = rsqrtf(red[0] / D + 1e-5f);
    for (int c = threadIdx.x; c < D; c += 256)
        hb[(size_t)blockIdx.x * D + c] = f2bf((xr[c] - mu) * rstd * g[c] + b[c]);
}

__global__ __launch_bounds__(256) void softmax_kernel(
    const float* __restrict__ S, unsigned short* __restrict__ Wout, int N, float scale)
{
    __shared__ float red[256];
    const float* sr = S + (size_t)blockIdx.x * N;
    float m = -3.4e38f;
    for (int c = threadIdx.x; c < N; c += 256) m = fmaxf(m, sr[c] * scale);
    red[threadIdx.x] = m; __syncthreads();
    for (int o = 128; o > 0; o >>= 1) {
        if (threadIdx.x < o) red[threadIdx.x] = fmaxf(red[threadIdx.x], red[threadIdx.x + o]);
        __syncthreads();
    }
    m = red[0]; __syncthreads();
    float s = 0.f;
    for (int c = threadIdx.x; c < N; c += 256) s += __expf(sr[c] * scale - m);
    red[threadIdx.x] = s; __syncthreads();
    for (int o = 128; o > 0; o >>= 1) {
        if (threadIdx.x < o) red[threadIdx.x] += red[threadIdx.x + o];
        __syncthreads();
    }
    float inv = 1.f / red[0];
    for (int c = threadIdx.x; c < N; c += 256)
        Wout[(size_t)blockIdx.x * N + c] = f2bf(__expf(sr[c] * scale - m) * inv);
}

__global__ void cvt_kernel(const float* __restrict__ in, unsigned short* __restrict__ out, int n) {
    int i = blockIdx.x * 256 + threadIdx.x;
    if (i < n) out[i] = f2bf(in[i]);
}

extern "C" void kernel_launch(void* const* d_in, const int* in_sizes, int n_in,
                              void* d_out, int out_size, void* d_ws, size_t ws_size,
                              hipStream_t stream) {
    const float* x      = (const float*)d_in[0];
    const float* norm_w = (const float*)d_in[1];
    const float* norm_b = (const float*)d_in[2];
    const float* qkv_w  = (const float*)d_in[3];
    const float* qkv_b  = (const float*)d_in[4];
    const float* proj_w = (const float*)d_in[5];
    const float* proj_b = (const float*)d_in[6];
    float* out = (float*)d_out;
    (void)in_sizes; (void)n_in; (void)out_size; (void)ws_size;

    const int Bn = 8, L = 4096, D = 1024;
    const int R = Bn * L;                           // 32768 rows

    char* ws = (char*)d_ws;
    size_t off = 0;
    auto take = [&](size_t bytes) {
        char* p = ws + off;
        off = (off + bytes + 255) & ~(size_t)255;
        return p;
    };
    unsigned short* hb   = (unsigned short*)take((size_t)R * D * 2);        // LN(x), bf16
    unsigned short* wq   = (unsigned short*)take((size_t)3 * D * D * 2);    // qkv_w, bf16
    unsigned short* wp   = (unsigned short*)take((size_t)D * D * 2);        // proj_w, bf16
    unsigned short* qkvb = (unsigned short*)take((size_t)R * 3 * D * 2);    // qkv, bf16
    float*          Sb   = (float*)take((size_t)Bn * D * D * 4);            // logits, f32
    unsigned short* wsm  = (unsigned short*)take((size_t)Bn * D * D * 2);   // softmax, bf16
    unsigned short* ab   = (unsigned short*)take((size_t)R * D * 2);        // attn out, bf16

    cvt_kernel<<<(3 * D * D + 255) / 256, 256, 0, stream>>>(qkv_w, wq, 3 * D * D);
    cvt_kernel<<<(D * D + 255) / 256, 256, 0, stream>>>(proj_w, wp, D * D);
    ln_kernel<<<R, 256, 0, stream>>>(x, norm_w, norm_b, hb, D);

    // qkv = h @ qkv_w^T + qkv_b                       (32768 x 3072, K=1024)
    gemm_bf16_kernel<<<dim3(3 * D / BN, R / BM, 1), 256, 0, stream>>>(
        hb, D, 0, 0,  wq, D, 0, 0,  qkv_b,  nullptr, 0,
        nullptr, qkvb, 3 * D, 0, D);

    // S[i,j] = sum_c q[c,i] k[c,j] per batch          (1024 x 1024, K=4096)
    gemm_bf16_kernel<<<dim3(D / BN, D / BM, Bn), 256, 0, stream>>>(
        qkvb,     3 * D, (long long)L * 3 * D, 1,
        qkvb + D, 3 * D, (long long)L * 3 * D, 1,
        nullptr,  nullptr, 0,
        Sb, nullptr, D, (long long)D * D, L);

    softmax_kernel<<<Bn * D, 256, 0, stream>>>(Sb, wsm, D, 1.0f / 64.0f);

    // a[c,q] = sum_k v[c,k] w[q,k] per batch          (4096 x 1024, K=1024)
    gemm_bf16_kernel<<<dim3(D / BN, L / BM, Bn), 256, 0, stream>>>(
        qkvb + 2 * D, 3 * D, (long long)L * 3 * D, 0,
        wsm,          D,     (long long)D * D,     0,
        nullptr, nullptr, 0,
        nullptr, ab, D, (long long)L * D, D);

    // out = a @ proj_w^T + proj_b + x                 (32768 x 1024, K=1024)
    gemm_bf16_kernel<<<dim3(D / BN, R / BM, 1), 256, 0, stream>>>(
        ab, D, 0, 0,  wp, D, 0, 0,  proj_b,  x, 0,
        out, nullptr, D, 0, D);
}
